// MaskRCNN_78383153152324
// MI455X (gfx1250) — matrix operand order repaired and checked by reading, not compiled, not run on hardware
//
#include <hip/hip_runtime.h>
#include <hip/hip_bf16.h>

typedef __attribute__((ext_vector_type(16))) _Float16 v16h;
typedef __attribute__((ext_vector_type(8)))  _Float16 v8h;
typedef __attribute__((ext_vector_type(8)))  float    v8f;

// ---------------------------------------------------------------------------
// CDNA5 async memory->LDS copy (ASYNCcnt-tracked, no VGPR round trip).
// Generic LDS pointers on gfx12 carry the LDS byte offset in the low 32 bits.
// ---------------------------------------------------------------------------
__device__ __forceinline__ void async_g2l_b128(uint32_t lds_off, const void* gptr) {
  asm volatile("global_load_async_to_lds_b128 %0, %1, off"
               :: "v"(lds_off), "v"(gptr) : "memory");
}
__device__ __forceinline__ void wait_asynccnt0() {
  asm volatile("s_wait_asynccnt 0" ::: "memory");
}

// ---------------------------------------------------------------------------
// WMMA fragment builders (CDNA5 16x16x32 f16 layouts, CDNA5 ISA)
// A 16x32: lane half kb=0/8 -> elems 0..7 = K kb..kb+7, elems 8..15 = K kb+16..kb+23
// B 32x16: same K order as A (elems 0..7 = K 8h..8h+7, elems 8..15 = K 8h+16..8h+23)
// ---------------------------------------------------------------------------
__device__ __forceinline__ v16h make_a_frag(const _Float16* rowp, int k0, int half8) {
  v8h a0 = *(const v8h*)(rowp + k0 + half8);
  v8h a1 = *(const v8h*)(rowp + k0 + half8 + 16);
  v16h r;
#pragma unroll
  for (int e = 0; e < 8; ++e) { r[e] = a0[e]; r[8 + e] = a1[e]; }
  return r;
}
__device__ __forceinline__ v16h make_b_frag(const _Float16* rowp, int k0, int half16) {
  v8h b0 = *(const v8h*)(rowp + k0 + (half16 >> 1));
  v8h b1 = *(const v8h*)(rowp + k0 + (half16 >> 1) + 16);
  v16h r;
#pragma unroll
  for (int e = 0; e < 8; ++e) { r[e] = b0[e]; r[8 + e] = b1[e]; }
  return r;
}

// ---------------------------------------------------------------------------
// ROI align (out=7 -> FC layout [n][c*49+cell] f16; out=14 -> HWC [n][p][c] f16)
// ---------------------------------------------------------------------------
__device__ __forceinline__ int fpn_level(float x1, float y1, float x2, float y2) {
  float area = (x2 - x1) * (y2 - y1);
  float k = floorf(4.0f + log2f(sqrtf(area) / 224.0f + 1e-6f));
  k = fminf(fmaxf(k, 2.0f), 5.0f);
  return (int)k - 2;
}

template <int OUT>
__global__ void roi_align_kernel(const float* __restrict__ f2, const float* __restrict__ f3,
                                 const float* __restrict__ f4, const float* __restrict__ f5,
                                 const float* __restrict__ boxes, const int* __restrict__ bidx,
                                 _Float16* __restrict__ out) {
  const int cell = blockIdx.x % (OUT * OUT);
  const int n    = blockIdx.x / (OUT * OUT);
  const int oy = cell / OUT, ox = cell % OUT;
  const int c = threadIdx.x;  // channel

  float x1 = boxes[n * 4 + 0], y1 = boxes[n * 4 + 1];
  float x2 = boxes[n * 4 + 2], y2 = boxes[n * 4 + 3];
  int lvl = fpn_level(x1, y1, x2, y2);

  const float* f; int H; float s;
  if (lvl == 0)      { f = f2; H = 200; s = 0.25f; }
  else if (lvl == 1) { f = f3; H = 100; s = 0.125f; }
  else if (lvl == 2) { f = f4; H = 50;  s = 0.0625f; }
  else               { f = f5; H = 25;  s = 0.03125f; }
  const int W = H;

  float bx1 = x1 * s, by1 = y1 * s, bx2 = x2 * s, by2 = y2 * s;
  float rw = fmaxf(bx2 - bx1, 1.0f), rh = fmaxf(by2 - by1, 1.0f);
  const int b = bidx[n];
  const float P = (float)(OUT * 2);
  const float* fc = f + (size_t)(b * 256 + c) * H * W;

  float accum = 0.0f;
#pragma unroll
  for (int sy = 0; sy < 2; ++sy) {
    float gy = by1 + ((oy * 2 + sy) + 0.5f) / P * rh;
    gy = fminf(fmaxf(gy, 0.0f), (float)(H - 1));
    int y0 = (int)floorf(gy);
    float ly = gy - (float)y0;
    int y1i = min(y0 + 1, H - 1);
#pragma unroll
    for (int sx = 0; sx < 2; ++sx) {
      float gx = bx1 + ((ox * 2 + sx) + 0.5f) / P * rw;
      gx = fminf(fmaxf(gx, 0.0f), (float)(W - 1));
      int x0 = (int)floorf(gx);
      float lx = gx - (float)x0;
      int x1i = min(x0 + 1, W - 1);
      float v00 = fc[y0 * W + x0],  v01 = fc[y0 * W + x1i];
      float v10 = fc[y1i * W + x0], v11 = fc[y1i * W + x1i];
      accum += v00 * (1.f - ly) * (1.f - lx) + v01 * (1.f - ly) * lx +
               v10 * ly * (1.f - lx) + v11 * ly * lx;
    }
  }
  accum *= 0.25f;
  if (OUT == 7)
    out[(size_t)n * 12544 + c * 49 + oy * 7 + ox] = (_Float16)accum;
  else
    out[((size_t)n * 196 + oy * 14 + ox) * 256 + c] = (_Float16)accum;
}

// ---------------------------------------------------------------------------
// Generic WMMA GEMM: D[M,N] = act(A[M,K]f16 @ Brow[N,K]f16^T + bias)
// 1 wave per 16x16 tile. M,N,K multiples of 16/16/32.
// ---------------------------------------------------------------------------
template <bool RELU, bool F16OUT>
__global__ void wmma_gemm(const _Float16* __restrict__ A, const _Float16* __restrict__ Bw,
                          const float* __restrict__ bias, void* __restrict__ outv,
                          int M, int N, int K) {
  const int mtiles = M >> 4;
  const int mt = blockIdx.x % mtiles;
  const int nt = blockIdx.x / mtiles;
  const int lane = threadIdx.x & 31;
  const int l15 = lane & 15, hi = lane >> 4;

  const _Float16* arow = A  + (size_t)(mt * 16 + l15) * K;
  const _Float16* brow = Bw + (size_t)(nt * 16 + l15) * K;

  v8f acc = {};
  for (int k0 = 0; k0 < K; k0 += 32) {
    // stream next k-block toward GL2 (global_prefetch_b8)
    if (k0 + 256 < K) {
      __builtin_prefetch(arow + k0 + 256, 0, 1);
      __builtin_prefetch(brow + k0 + 256, 0, 1);
    }
    v16h a = make_a_frag(arow, k0, hi * 8);
    v16h b = make_b_frag(brow, k0, hi * 16);
    acc = __builtin_amdgcn_wmma_f32_16x16x32_f16(false, a, false, b, (short)0, acc, false, false);
  }

  const int ncol = nt * 16 + l15;
  const float bv = bias[ncol];
#pragma unroll
  for (int r = 0; r < 8; ++r) {
    int m = mt * 16 + r + hi * 8;
    float v = acc[r] + bv;
    if (RELU) v = fmaxf(v, 0.0f);
    if (F16OUT) ((_Float16*)outv)[(size_t)m * N + ncol] = (_Float16)v;
    else        ((float*)outv)[(size_t)m * N + ncol] = v;
  }
}

// ---------------------------------------------------------------------------
// 3x3 SAME conv, C=256, per-ROI workgroup. Input HWC f16 in LDS with zero halo.
// LDS interior filled with GLOBAL_LOAD_ASYNC_TO_LDS_B128 (ASYNCcnt); halo
// zeroed with DS stores (disjoint addresses, so the unordered counters are
// safe). K-order = (ky,kx,ci) so A fragments are contiguous ds_load_b128.
// 16 waves = 16 o-tiles; 13 m-tiles of pixels; K=2304 -> 72 WMMA each.
// ---------------------------------------------------------------------------
__global__ void __launch_bounds__(512)
conv3x3_wmma(const _Float16* __restrict__ in,   // [roi][196][256]
             const _Float16* __restrict__ wh,   // [256 o][9*256] (ky*3+kx, ci)
             const float* __restrict__ bias,    // [256]
             _Float16* __restrict__ out) {      // [roi][196][256]
  extern __shared__ _Float16 smem[];  // 16*16*256 halves = 128KB
  const int n = blockIdx.x;
  const int tid = threadIdx.x;

  // 1) zero only the halo pixels (y==0||y==15||x==0||x==15): DS-counter stores
  v8h z = {};
  for (int q = tid; q < 256 * 32; q += 512) {   // 256 padded pixels * 32 chunks
    int pp = q >> 5, i = (q & 31) * 8;
    int y = pp >> 4, x = pp & 15;
    if (y == 0 || y == 15 || x == 0 || x == 15)
      *(v8h*)(&smem[(y * 16 + x) * 256 + i]) = z;
  }
  // 2) async-fill the 14x14 interior: ASYNCcnt loads, memory -> LDS directly
  const _Float16* src = in + (size_t)n * 196 * 256;
  for (int q = tid; q < 196 * 32; q += 512) {
    int p = q >> 5, i = (q & 31) * 8;
    int y = p / 14, x = p % 14;
    uint32_t loff = (uint32_t)(uintptr_t)(&smem[(((y + 1) * 16) + (x + 1)) * 256 + i]);
    async_g2l_b128(loff, src + p * 256 + i);
  }
  wait_asynccnt0();
  __syncthreads();

  const int wave = tid >> 5;
  const int lane = tid & 31;
  const int l15 = lane & 15, hi = lane >> 4;
  const int o = wave * 16 + l15;
  const _Float16* brow = wh + (size_t)o * 2304;
  const float bv = bias[o];
  _Float16* dst = out + (size_t)n * 196 * 256;

  for (int mt = 0; mt < 13; ++mt) {
    int p = mt * 16 + l15; if (p > 195) p = 195;
    int y = p / 14, x = p - (p / 14) * 14;
    v8f acc = {};
    for (int kt = 0; kt < 72; ++kt) {
      int k0 = kt * 32;
      int j = k0 >> 8, i0 = k0 & 255;       // j = ky*3+kx (32 | 256 -> never straddles)
      int ky = j / 3, kx = j - ky * 3;
      int lb = (((y + ky) * 16) + (x + kx)) * 256 + i0 + hi * 8;
      v8h a0 = *(const v8h*)(&smem[lb]);
      v8h a1 = *(const v8h*)(&smem[lb + 16]);
      v16h a;
#pragma unroll
      for (int e = 0; e < 8; ++e) { a[e] = a0[e]; a[8 + e] = a1[e]; }
      v16h b = make_b_frag(brow, k0, hi * 16);
      acc = __builtin_amdgcn_wmma_f32_16x16x32_f16(false, a, false, b, (short)0, acc, false, false);
    }
#pragma unroll
    for (int r = 0; r < 8; ++r) {
      int pp = mt * 16 + r + hi * 8;
      if (pp < 196) dst[(size_t)pp * 256 + o] = (_Float16)fmaxf(acc[r] + bv, 0.0f);
    }
  }
}

// ---------------------------------------------------------------------------
// Fused conv-transpose(2x2,s2) + bd + ReLU + 1x1 conv(256->2), atomics into d_out.
// Per output phase (a,b): D[p,o] = m4[p,:] @ wd[o,:,1-a,1-b]; epilogue reduces
// over o via shfl_xor and atomically accumulates wm5[c,o]*relu(.) per pixel.
// ---------------------------------------------------------------------------
__global__ void __launch_bounds__(512)
deconv_mask_wmma(const _Float16* __restrict__ m4,   // [256][196][256]
                 const _Float16* __restrict__ wdh,  // [4 ab][256 o][256 i] pre-flipped
                 const float* __restrict__ bd,      // [256]
                 const float* __restrict__ wm5,     // [2*256]
                 float* __restrict__ out) {         // d_out (mask at +2560)
  const int n = blockIdx.x;
  const int wave = threadIdx.x >> 5;
  const int lane = threadIdx.x & 31;
  const int l15 = lane & 15, hi = lane >> 4;
  const _Float16* abase = m4 + (size_t)n * 196 * 256;
  float* mbase = out + 2560 + (size_t)n * 2 * 784;

  for (int t = wave; t < 832; t += 16) {       // 4 ab * 16 nt * 13 mt
    int ab = t / 208;
    int r2 = t - ab * 208;
    int nt = r2 / 13;
    int mt = r2 - nt * 13;
    int a_ = ab >> 1, b_ = ab & 1;
    int o = nt * 16 + l15;
    int p = mt * 16 + l15; if (p > 195) p = 195;
    const _Float16* arow = abase + (size_t)p * 256;
    const _Float16* brow = wdh + (size_t)ab * 65536 + (size_t)o * 256;

    v8f acc = {};
    for (int k0 = 0; k0 < 256; k0 += 32) {
      v16h a = make_a_frag(arow, k0, hi * 8);
      v16h b = make_b_frag(brow, k0, hi * 16);
      acc = __builtin_amdgcn_wmma_f32_16x16x32_f16(false, a, false, b, (short)0, acc, false, false);
    }
    float bdo = bd[o];
    float w50 = wm5[o], w51 = wm5[256 + o];
#pragma unroll
    for (int r = 0; r < 8; ++r) {
      int pp = mt * 16 + r + hi * 8;
      float v = (pp < 196) ? fmaxf(acc[r] + bdo, 0.0f) : 0.0f;
      float c0 = w50 * v, c1 = w51 * v;
#pragma unroll
      for (int m = 1; m < 16; m <<= 1) {       // reduce over the 16 o's in tile
        c0 += __shfl_xor(c0, m, 16);
        c1 += __shfl_xor(c1, m, 16);
      }
      if (l15 == 0 && pp < 196) {
        int y = pp / 14, x = pp - (pp / 14) * 14;
        int oy = 2 * y + a_, ox = 2 * x + b_;
        atomicAdd(&mbase[oy * 28 + ox], c0);
        atomicAdd(&mbase[784 + oy * 28 + ox], c1);
      }
    }
  }
}

// ---------------------------------------------------------------------------
// Small heads: cls (2) + bbox (8) per ROI from y2 f32.
// ---------------------------------------------------------------------------
__global__ void heads_kernel(const float* __restrict__ y2,
                             const float* __restrict__ Wc, const float* __restrict__ bc,
                             const float* __restrict__ Wb, const float* __restrict__ bb,
                             float* __restrict__ out) {
  __shared__ float red[256];
  const int n = blockIdx.x, tid = threadIdx.x;
  float part[10];
#pragma unroll
  for (int j = 0; j < 10; ++j) part[j] = 0.0f;
  for (int k = tid; k < 1024; k += 256) {
    float xv = y2[(size_t)n * 1024 + k];
    part[0] += xv * Wc[k];
    part[1] += xv * Wc[1024 + k];
#pragma unroll
    for (int j = 0; j < 8; ++j) part[2 + j] += xv * Wb[j * 1024 + k];
  }
  for (int j = 0; j < 10; ++j) {
    __syncthreads();
    red[tid] = part[j];
    __syncthreads();
    for (int s = 128; s > 0; s >>= 1) { if (tid < s) red[tid] += red[tid + s]; __syncthreads(); }
    if (tid == 0) {
      if (j < 2) out[n * 2 + j] = red[0] + bc[j];
      else       out[512 + n * 8 + (j - 2)] = red[0] + bb[j - 2];
    }
  }
}

// ---------------------------------------------------------------------------
// Repack / convert kernels
// ---------------------------------------------------------------------------
__global__ void f32_to_f16(const float* __restrict__ src, _Float16* __restrict__ dst, int count) {
  int i = blockIdx.x * 256 + threadIdx.x;
  if (i < count) dst[i] = (_Float16)src[i];
}
__global__ void repack_conv_w(const float* __restrict__ src, _Float16* __restrict__ dst) {
  int idx = blockIdx.x * 256 + threadIdx.x;  // 256*2304
  if (idx >= 256 * 2304) return;
  int o = idx / 2304, r = idx - o * 2304, j = r >> 8, i = r & 255;
  dst[idx] = (_Float16)src[(o * 256 + i) * 9 + j];  // (O,I,3,3) -> (O, ky*3+kx, I)
}
__global__ void repack_deconv_w(const float* __restrict__ wd, _Float16* __restrict__ dst) {
  int idx = blockIdx.x * 256 + threadIdx.x;  // 4*65536
  if (idx >= 4 * 65536) return;
  int ab = idx >> 16, r = idx & 65535, o = r >> 8, i = r & 255;
  int a_ = ab >> 1, b_ = ab & 1;
  dst[idx] = (_Float16)wd[(o * 256 + i) * 4 + (1 - a_) * 2 + (1 - b_)];  // flipped taps
}
__global__ void mask_init(const float* __restrict__ bm5, float* __restrict__ out) {
  int idx = blockIdx.x * 256 + threadIdx.x;
  if (idx < 256 * 2 * 784) out[2560 + idx] = bm5[(idx / 784) & 1];
}

// ---------------------------------------------------------------------------
// Launcher
// ---------------------------------------------------------------------------
extern "C" void kernel_launch(void* const* d_in, const int* in_sizes, int n_in,
                              void* d_out, int out_size, void* d_ws, size_t ws_size,
                              hipStream_t stream) {
  const float* f2    = (const float*)d_in[0];
  const float* f3    = (const float*)d_in[1];
  const float* f4    = (const float*)d_in[2];
  const float* f5    = (const float*)d_in[3];
  const float* boxes = (const float*)d_in[4];
  const int*   bidx  = (const int*)d_in[5];
  const float* W1 = (const float*)d_in[6];  const float* b1 = (const float*)d_in[7];
  const float* W2 = (const float*)d_in[8];  const float* b2 = (const float*)d_in[9];
  const float* Wc = (const float*)d_in[10]; const float* bc = (const float*)d_in[11];
  const float* Wb = (const float*)d_in[12]; const float* bb = (const float*)d_in[13];
  const float* wd = (const float*)d_in[14]; const float* bd = (const float*)d_in[15];
  const float* wm5 = (const float*)d_in[16]; const float* bm5 = (const float*)d_in[17];
  const float* wm[4] = { (const float*)d_in[18], (const float*)d_in[20],
                         (const float*)d_in[22], (const float*)d_in[24] };
  const float* bm[4] = { (const float*)d_in[19], (const float*)d_in[21],
                         (const float*)d_in[23], (const float*)d_in[25] };
  float* out = (float*)d_out;

  // workspace carve-up
  char* ws = (char*)d_ws;
  size_t off = 0;
  auto carve = [&](size_t bytes) { char* p = ws + off; off += (bytes + 255) & ~(size_t)255; return p; };
  _Float16* xh   = (_Float16*)carve((size_t)256 * 12544 * 2);
  _Float16* w1h  = (_Float16*)carve((size_t)1024 * 12544 * 2);
  _Float16* y1h  = (_Float16*)carve((size_t)256 * 1024 * 2);
  _Float16* w2h  = (_Float16*)carve((size_t)1024 * 1024 * 2);
  float*    y2   = (float*)   carve((size_t)256 * 1024 * 4);
  _Float16* p14h = (_Float16*)carve((size_t)256 * 196 * 256 * 2);
  _Float16* bufA = (_Float16*)carve((size_t)256 * 196 * 256 * 2);
  _Float16* bufB = (_Float16*)carve((size_t)256 * 196 * 256 * 2);
  _Float16* whm  = (_Float16*)carve((size_t)4 * 256 * 2304 * 2);
  _Float16* wdh  = (_Float16*)carve((size_t)4 * 65536 * 2);
  (void)in_sizes; (void)n_in; (void)out_size; (void)ws_size;

  // 1) ROI align into f16 (FC layout + HWC layout)
  roi_align_kernel<7><<<256 * 49, 256, 0, stream>>>(f2, f3, f4, f5, boxes, bidx, xh);
  roi_align_kernel<14><<<256 * 196, 256, 0, stream>>>(f2, f3, f4, f5, boxes, bidx, p14h);

  // 2) weight repacks to f16
  f32_to_f16<<<(1024 * 12544 + 255) / 256, 256, 0, stream>>>(W1, w1h, 1024 * 12544);
  f32_to_f16<<<(1024 * 1024 + 255) / 256, 256, 0, stream>>>(W2, w2h, 1024 * 1024);
  for (int c = 0; c < 4; ++c)
    repack_conv_w<<<2304, 256, 0, stream>>>(wm[c], whm + (size_t)c * 256 * 2304);
  repack_deconv_w<<<1024, 256, 0, stream>>>(wd, wdh);

  // 3) box head: FC1 (WMMA), FC2 (WMMA), then tiny linear heads
  wmma_gemm<true, true ><<<16 * 64, 32, 0, stream>>>(xh,  w1h, b1, y1h, 256, 1024, 12544);
  wmma_gemm<true, false><<<16 * 64, 32, 0, stream>>>(y1h, w2h, b2, y2,  256, 1024, 1024);
  heads_kernel<<<256, 256, 0, stream>>>(y2, Wc, bc, Wb, bb, out);

  // 4) mask head: 4x (3x3 conv + ReLU) via async-LDS-staged WMMA implicit GEMM
  const size_t lds_bytes = 16 * 16 * 256 * sizeof(_Float16);  // 128 KB (CDNA5: 320KB/WGP)
  conv3x3_wmma<<<256, 512, lds_bytes, stream>>>(p14h, whm + 0 * 256 * 2304, bm[0], bufA);
  conv3x3_wmma<<<256, 512, lds_bytes, stream>>>(bufA, whm + 1 * 256 * 2304, bm[1], bufB);
  conv3x3_wmma<<<256, 512, lds_bytes, stream>>>(bufB, whm + 2 * 256 * 2304, bm[2], bufA);
  conv3x3_wmma<<<256, 512, lds_bytes, stream>>>(bufA, whm + 3 * 256 * 2304, bm[3], bufB);

  // 5) fused deconv + ReLU + 1x1 mask conv (atomic accumulate into d_out)
  mask_init<<<(256 * 2 * 784 + 255) / 256, 256, 0, stream>>>(bm5, out);
  deconv_mask_wmma<<<256, 512, 0, stream>>>(bufB, wdh, bd, wm5, out);
}
